// TransformerBlock_25366076850665
// MI455X (gfx1250) — matrix operand order repaired and checked by reading, not compiled, not run on hardware
//
#include <hip/hip_runtime.h>
#include <hip/hip_bf16.h>

// ---------------------------------------------------------------------------
// Fused xLSTM-transformer block for gfx1250 (MI455X), wave32 + WMMA f16.
// Round 4: B fragments of each K-chunk preloaded into distinct registers
// before the WMMA chain so the scheduler can issue partial s_wait_loadcnt
// and overlap fragment loads with WMMA execution inside a wave.
// ---------------------------------------------------------------------------

typedef __attribute__((ext_vector_type(16))) _Float16 v16h;
typedef __attribute__((ext_vector_type(8)))  float    v8f;

constexpr int BATCH = 16384;
constexpr int SEQL  = 8;
constexpr int PAT   = 4;
constexpr int DIM   = 140;   // token dim / d_k
constexpr int DIMP  = 160;   // padded token dim: 5 K-chunks of 32, 10 N-tiles of 16
constexpr int FF    = 1024;  // dff
constexpr int ROWS  = BATCH * PAT;   // 65536 rows for h / z2 / z5
constexpr int CROWS = BATCH * SEQL;  // 131072 rows for cat / q / k / v

__device__ inline v8f wmma16(v16h a, v16h b, v8f c) {
  return __builtin_amdgcn_wmma_f32_16x16x32_f16(
      false, a, false, b, (short)0, c, false, false);
}

// ---- fragment access -------------------------------------------------------

// A: 16x32 f16 (MxK) fragment loaded directly from global row-major memory.
// Per ISA 7.12.2 the per-lane data is two contiguous 8-half runs:
//   elements 0..7  = K = hi..hi+7,    elements 8..15 = K = hi+16..hi+23,
// where hi = (lane>=16)*8 and row M = lane%16.  -> two global_load_b128.
__device__ inline v16h load_a_global(const _Float16* gTile, int ldg) {
  int lane = threadIdx.x & 31;
  int m  = lane & 15;
  int hi = (lane >> 4) * 8;
  const _Float16* p = gTile + (size_t)m * ldg + hi;
  v16h a;
  *(uint4*)&a       = *(const uint4*)p;         // K hi .. hi+7
  *((uint4*)&a + 1) = *(const uint4*)(p + 16);  // K hi+16 .. hi+23
  return a;
}

// B fragments live pre-swizzled in GLOBAL memory, fragment-major:
//   Wp[((kc*NT + nt)*32 + lane)*16 + e] = W[kc*32 + (lane>=16)*8 + (e&7) + (e>=8)*16][nt*16 + lane%16]
// A lane's whole fragment is 32 contiguous, 32B-aligned bytes -> 2x b128.
__device__ inline v16h load_bp(const _Float16* Wp, int kc, int NT, int nt) {
  int lane = threadIdx.x & 31;
  const _Float16* p = Wp + ((((size_t)kc * NT + nt) * 32 + lane) << 4);
  v16h b;
  *(uint4*)&b       = *(const uint4*)p;
  *((uint4*)&b + 1) = *(const uint4*)(p + 8);
  return b;
}

// ---------------------------------------------------------------------------
// One-time weight prep: f32 row-major [Kv x Nv] -> f16 fragment-major
// padded [Kp x Np] (Kp=32*ceil, Np=16*NT), zero-padded. grid.x*256 == Kp*Np.
// ---------------------------------------------------------------------------
__global__ __launch_bounds__(256)
void k_prep_w(const float* __restrict__ W, _Float16* __restrict__ Wp,
              int NT, int Kv, int Nv) {
  size_t i = (size_t)blockIdx.x * 256 + threadIdx.x;
  int within = (int)(i & 511);          // lane*16 + e
  size_t blk = i >> 9;                  // (kc*NT + nt)
  int nt = (int)(blk % NT);
  int kc = (int)(blk / NT);
  int lane = within >> 4, e = within & 15;
  int gk = kc * 32 + ((lane >> 4) << 3) + (e & 7) + ((e >> 3) << 4);
  int gc = nt * 16 + (lane & 15);
  float v = (gk < Kv && gc < Nv) ? W[(size_t)gk * Nv + gc] : 0.f;
  Wp[i] = (_Float16)v;
}

__global__ __launch_bounds__(256)
void k_prep_bias(const float* __restrict__ b, const float* __restrict__ rb,
                 float* __restrict__ out) {
  int i = blockIdx.x * 256 + threadIdx.x;   // FF elements
  out[i] = b[i] + rb[i];
}

// ---------------------------------------------------------------------------
// Kernel 1: hc = h @ Whc + bhc ; build padded f16 cat = [s_i ; hc]  [CROWS,160]
// grid: ROWS/64. M=64, N=160 (10 tiles), K=1024. wave -> mt=wave&3, 5 nt's.
// ---------------------------------------------------------------------------
__global__ __launch_bounds__(256)
void k_hc_cat(const _Float16* __restrict__ hF, const _Float16* __restrict__ WhcP,
              const float* __restrict__ bhc, const float* __restrict__ state,
              int t, _Float16* __restrict__ cat) {
  int rowBase = blockIdx.x * 64;
  int wave = threadIdx.x >> 5;
  int mt = wave & 3, ntBase = wave >> 2;
  v8f acc[5] = {};
  for (int kc = 0; kc < FF / 32; ++kc) {
    v16h a = load_a_global(hF + (size_t)(rowBase + mt * 16) * FF + kc * 32, FF);
    v16h bb[5];
#pragma unroll
    for (int j = 0; j < 5; ++j) bb[j] = load_bp(WhcP, kc, 10, ntBase + 2 * j);
#pragma unroll
    for (int j = 0; j < 5; ++j) acc[j] = wmma16(a, bb[j], acc[j]);
  }
  int lane = threadIdx.x & 31, nL = lane & 15, mB = (lane >> 4) * 8;
#pragma unroll
  for (int j = 0; j < 5; ++j) {
    int nt = ntBase + 2 * j;
#pragma unroll
    for (int r = 0; r < 8; ++r) {
      int row = rowBase + mt * 16 + mB + r;   // h row
      int col = nt * 16 + nL;                 // 0..159
      int b = row >> 2, p = row & 3;
      size_t cr = ((size_t)b * 8 + 4 + p) * DIMP;  // hc rows of cat
      float v = (col < DIM) ? acc[j][r] + bhc[col] : 0.f;
      cat[cr + col] = (_Float16)v;
    }
  }
  // s_i rows (f32 -> f16, zero-padded cols)
  int bb2 = rowBase >> 2;  // 16 batches per block
  for (int i = threadIdx.x; i < 16 * PAT * DIMP; i += 256) {
    int c = i % DIMP, pr = i / DIMP, p = pr & 3, bl = pr >> 2;
    int bi2 = bb2 + bl;
    float v = 0.f;
    if (c < DIM) v = state[((((size_t)bi2 * SEQL) + t) * PAT + p) * DIM + c];
    cat[((size_t)bi2 * 8 + p) * DIMP + c] = (_Float16)v;
  }
}

// ---------------------------------------------------------------------------
// Kernel 2: q/k/v = cat @ W + b  (grid.y selects projection)
// grid: (CROWS/64, 3). M=64, N=160, K=160.
// ---------------------------------------------------------------------------
__global__ __launch_bounds__(256)
void k_qkv(const _Float16* __restrict__ cat,
           const _Float16* WqP, const float* bq,
           const _Float16* WkP, const float* bk,
           const _Float16* WvP, const float* bv,
           _Float16* qf, _Float16* kf, _Float16* vf) {
  const _Float16* Wp; const float* bias; _Float16* out;
  if (blockIdx.y == 0)      { Wp = WqP; bias = bq; out = qf; }
  else if (blockIdx.y == 1) { Wp = WkP; bias = bk; out = kf; }
  else                      { Wp = WvP; bias = bv; out = vf; }
  int rowBase = blockIdx.x * 64;
  int wave = threadIdx.x >> 5;
  int mt = wave & 3, ntBase = wave >> 2;
  v8f acc[5] = {};
#pragma unroll
  for (int kc = 0; kc < DIMP / 32; ++kc) {
    v16h a = load_a_global(cat + (size_t)(rowBase + mt * 16) * DIMP + kc * 32, DIMP);
    v16h bb[5];
#pragma unroll
    for (int j = 0; j < 5; ++j) bb[j] = load_bp(Wp, kc, 10, ntBase + 2 * j);
#pragma unroll
    for (int j = 0; j < 5; ++j) acc[j] = wmma16(a, bb[j], acc[j]);
  }
  int lane = threadIdx.x & 31, nL = lane & 15, mB = (lane >> 4) * 8;
#pragma unroll
  for (int j = 0; j < 5; ++j) {
    int nt = ntBase + 2 * j;
#pragma unroll
    for (int r = 0; r < 8; ++r) {
      int row = rowBase + mt * 16 + mB + r;
      int col = nt * 16 + nL;
      float v = (col < DIM) ? acc[j][r] + bias[col] : 0.f;
      out[(size_t)row * DIMP + col] = (_Float16)v;
    }
  }
}

// ---------------------------------------------------------------------------
// Kernel 3: attention (only the 4 live query rows), z1 = s_i + attn, LN1 -> z2
// One wave per batch element; 8 batches per block. grid: BATCH/8.
// ---------------------------------------------------------------------------
__global__ __launch_bounds__(256)
void k_attn_ln(const _Float16* __restrict__ qf, const _Float16* __restrict__ kf,
               const _Float16* __restrict__ vf, const float* __restrict__ state,
               int t, const float* g1, const float* be1,
               _Float16* __restrict__ z2) {
  __shared__ float ps[8][4][8];
  __shared__ float z1s[8][4][DIM];
  int wave = threadIdx.x >> 5, lane = threadIdx.x & 31;
  int b = blockIdx.x * 8 + wave;
  const _Float16* qb = qf + (size_t)b * 8 * DIMP;
  const _Float16* kb = kf + (size_t)b * 8 * DIMP;
  const _Float16* vb = vf + (size_t)b * 8 * DIMP;
  int qr = lane >> 3, kr = lane & 7;
  float s = 0.f;
  for (int d = 0; d < DIM; ++d)
    s += (float)qb[qr * DIMP + d] * (float)kb[kr * DIMP + d];
  s *= rsqrtf((float)DIM);
  float mx = s;
  for (int o = 4; o; o >>= 1) mx = fmaxf(mx, __shfl_xor(mx, o, 8));
  float e = __expf(s - mx);
  float sm = e;
  for (int o = 4; o; o >>= 1) sm += __shfl_xor(sm, o, 8);
  ps[wave][qr][kr] = e / sm;   // same-wave producer/consumer: in-order
  for (int q2 = 0; q2 < PAT; ++q2) {
    for (int d = lane; d < DIM; d += 32) {
      float a = 0.f;
#pragma unroll
      for (int k2 = 0; k2 < 8; ++k2)
        a += ps[wave][q2][k2] * (float)vb[k2 * DIMP + d];
      z1s[wave][q2][d] =
          state[((((size_t)b * SEQL) + t) * PAT + q2) * DIM + d] + a;
    }
  }
  for (int q2 = 0; q2 < PAT; ++q2) {
    float s1 = 0.f, s2 = 0.f;
    for (int d = lane; d < DIM; d += 32) {
      float v = z1s[wave][q2][d];
      s1 += v; s2 += v * v;
    }
    for (int o = 16; o; o >>= 1) { s1 += __shfl_xor(s1, o, 32); s2 += __shfl_xor(s2, o, 32); }
    float mu = s1 / (float)DIM;
    float rstd = rsqrtf(s2 / (float)DIM - mu * mu + 1e-5f);
    size_t base = ((size_t)b * PAT + q2) * DIMP;
    for (int d = lane; d < DIMP; d += 32) {
      _Float16 o16 = (_Float16)0.f;
      if (d < DIM)
        o16 = (_Float16)((z1s[wave][q2][d] - mu) * rstd * g1[d] + be1[d]);
      z2[base + d] = o16;
    }
  }
}

// ---------------------------------------------------------------------------
// Kernel 4: a1 = relu(z2 @ W1 + b1)   grid: (ROWS/64, FF/64). M=64,N=64,K=160.
// ---------------------------------------------------------------------------
__global__ __launch_bounds__(256)
void k_ffn1(const _Float16* __restrict__ z2, const _Float16* __restrict__ W1P,
            const float* __restrict__ b1, _Float16* __restrict__ a1) {
  int rowBase = blockIdx.x * 64, colBase = blockIdx.y * 64;
  int wave = threadIdx.x >> 5;
  int mt = wave & 3;
  int ntg0 = (colBase >> 4) + (wave >> 2);
  v8f acc[2] = {};
#pragma unroll
  for (int kc = 0; kc < DIMP / 32; ++kc) {
    v16h a = load_a_global(z2 + (size_t)(rowBase + mt * 16) * DIMP + kc * 32, DIMP);
    v16h bb[2];
#pragma unroll
    for (int j = 0; j < 2; ++j) bb[j] = load_bp(W1P, kc, FF / 16, ntg0 + 2 * j);
#pragma unroll
    for (int j = 0; j < 2; ++j) acc[j] = wmma16(a, bb[j], acc[j]);
  }
  int lane = threadIdx.x & 31, nL = lane & 15, mB = (lane >> 4) * 8;
#pragma unroll
  for (int j = 0; j < 2; ++j)
#pragma unroll
    for (int r = 0; r < 8; ++r) {
      int row = rowBase + mt * 16 + mB + r;
      int col = (ntg0 + 2 * j) * 16 + nL;
      float v = acc[j][r] + b1[col];
      a1[(size_t)row * FF + col] = (_Float16)fmaxf(v, 0.f);
    }
}

// ---------------------------------------------------------------------------
// Kernel 5: z4 = a1 @ W2 + b2 ; z5 = LN2(z2 + z4)  (block owns full rows)
// grid: ROWS/64. M=64, N=160, K=1024.
// ---------------------------------------------------------------------------
__global__ __launch_bounds__(256)
void k_ffn2_ln(const _Float16* __restrict__ a1, const _Float16* __restrict__ W2P,
               const float* __restrict__ b2, const _Float16* __restrict__ z2,
               const float* g2, const float* be2, _Float16* __restrict__ z5) {
  __shared__ float tbuf[64][DIM];
  __shared__ float stats[64][2];
  int rowBase = blockIdx.x * 64;
  int wave = threadIdx.x >> 5;
  int mt = wave & 3, ntBase = wave >> 2;
  v8f acc[5] = {};
  for (int kc = 0; kc < FF / 32; ++kc) {
    v16h a = load_a_global(a1 + (size_t)(rowBase + mt * 16) * FF + kc * 32, FF);
    v16h bb[5];
#pragma unroll
    for (int j = 0; j < 5; ++j) bb[j] = load_bp(W2P, kc, 10, ntBase + 2 * j);
#pragma unroll
    for (int j = 0; j < 5; ++j) acc[j] = wmma16(a, bb[j], acc[j]);
  }
  int lane = threadIdx.x & 31, nL = lane & 15, mB = (lane >> 4) * 8;
#pragma unroll
  for (int j = 0; j < 5; ++j) {
    int nt = ntBase + 2 * j;
#pragma unroll
    for (int r = 0; r < 8; ++r) {
      int lrow = mt * 16 + mB + r;
      int col = nt * 16 + nL;
      if (col < DIM) {
        float z4 = acc[j][r] + b2[col];
        float z2v = (float)z2[(size_t)(rowBase + lrow) * DIMP + col];
        tbuf[lrow][col] = z2v + z4;
      }
    }
  }
  __syncthreads();
  if (threadIdx.x < 64) {
    float s1 = 0.f, s2 = 0.f;
    for (int c = 0; c < DIM; ++c) { float v = tbuf[threadIdx.x][c]; s1 += v; s2 += v * v; }
    float mu = s1 / (float)DIM;
    stats[threadIdx.x][0] = mu;
    stats[threadIdx.x][1] = rsqrtf(s2 / (float)DIM - mu * mu + 1e-5f);
  }
  __syncthreads();
  for (int i = threadIdx.x; i < 64 * DIMP; i += 256) {
    int r = i / DIMP, c = i % DIMP;
    _Float16 o = (_Float16)0.f;
    if (c < DIM)
      o = (_Float16)((tbuf[r][c] - stats[r][0]) * stats[r][1] * g2[c] + be2[c]);
    z5[(size_t)(rowBase + r) * DIMP + c] = o;
  }
}

// ---------------------------------------------------------------------------
// Kernel 6 (dominant): all 4 gates G = h@Wg + z5@Rg + bsum on one 64x64 tile
// (K = 32 h-chunks + 5 padded z5-chunks), then fused exp-gated LSTM update.
// grid: (ROWS/64, FF/64). No LDS, no barriers: pure b128 + WMMA streaming.
// ---------------------------------------------------------------------------
__global__ __launch_bounds__(256)
void k_gates(const _Float16* __restrict__ hF, const _Float16* __restrict__ z5,
             const _Float16* __restrict__ WiP, const _Float16* __restrict__ WfP,
             const _Float16* __restrict__ WoP, const _Float16* __restrict__ WzP,
             const _Float16* __restrict__ RiP, const _Float16* __restrict__ RfP,
             const _Float16* __restrict__ RoP, const _Float16* __restrict__ RzP,
             const float* __restrict__ bsum,  // [4][FF], b_g + rb_g
             const float* __restrict__ mask, int t,
             float* __restrict__ Cst, float* __restrict__ Mst,
             float* __restrict__ Nst, float* __restrict__ H,
             _Float16* __restrict__ Hh) {
  int rowBase = blockIdx.x * 64, colBase = blockIdx.y * 64;
  int wave = threadIdx.x >> 5;
  int mt = wave & 3;
  int ntg0 = (colBase >> 4) + (wave >> 2);
  const _Float16* Wp[4] = {WiP, WfP, WoP, WzP};
  const _Float16* Rp[4] = {RiP, RfP, RoP, RzP};
  v8f acc[4][2] = {};
  for (int kc = 0; kc < 32; ++kc) {            // h @ Wg
    v16h a = load_a_global(hF + (size_t)(rowBase + mt * 16) * FF + kc * 32, FF);
    v16h bb[4][2];
#pragma unroll
    for (int g = 0; g < 4; ++g)
#pragma unroll
      for (int j = 0; j < 2; ++j)
        bb[g][j] = load_bp(Wp[g], kc, FF / 16, ntg0 + 2 * j);
#pragma unroll
    for (int g = 0; g < 4; ++g)
#pragma unroll
      for (int j = 0; j < 2; ++j)
        acc[g][j] = wmma16(a, bb[g][j], acc[g][j]);
  }
#pragma unroll
  for (int kc = 0; kc < 5; ++kc) {             // z5 @ Rg (K padded to 160)
    v16h a = load_a_global(z5 + (size_t)(rowBase + mt * 16) * DIMP + kc * 32, DIMP);
    v16h bb[4][2];
#pragma unroll
    for (int g = 0; g < 4; ++g)
#pragma unroll
      for (int j = 0; j < 2; ++j)
        bb[g][j] = load_bp(Rp[g], kc, FF / 16, ntg0 + 2 * j);
#pragma unroll
    for (int g = 0; g < 4; ++g)
#pragma unroll
      for (int j = 0; j < 2; ++j)
        acc[g][j] = wmma16(a, bb[g][j], acc[g][j]);
  }
  // fused state-update epilogue
  int lane = threadIdx.x & 31, nL = lane & 15, mB = (lane >> 4) * 8;
#pragma unroll
  for (int j = 0; j < 2; ++j)
#pragma unroll
    for (int r = 0; r < 8; ++r) {
      int row = rowBase + mt * 16 + mB + r;
      int col = (ntg0 + 2 * j) * 16 + nL;
      size_t idx = (size_t)row * FF + col;
      int bb2 = row >> 2, p = row & 3;
      float mk = mask[(((size_t)bb2 * SEQL) + t) * PAT + p];
      float it = acc[0][j][r] + bsum[0 * FF + col];
      float ft = acc[1][j][r] + bsum[1 * FF + col];
      float ot = acc[2][j][r] + bsum[2 * FF + col];
      float zt = acc[3][j][r] + bsum[3 * FF + col];
      float mprev = Mst[idx];
      float m_t = fmaxf(ft + mprev, it);
      float ie = __expf(it - m_t);
      float fe = __expf(ft + mprev - m_t);
      float og = 1.f / (1.f + __expf(-ot));
      float zg = tanhf(zt);
      float nprev = Nst[idx], cprev = Cst[idx], hprev = H[idx];
      float n_t = fe * nprev + ie;
      float c_t = (cprev * fe + zg * ie) * mk + (1.f - mk) * cprev;
      float h_t = og * (c_t / n_t) * mk + (1.f - mk) * hprev;
      Cst[idx] = c_t; Mst[idx] = m_t; Nst[idx] = n_t;
      H[idx] = h_t; Hh[idx] = (_Float16)h_t;
    }
}

// ---------------------------------------------------------------------------
extern "C" void kernel_launch(void* const* d_in, const int* in_sizes, int n_in,
                              void* d_out, int out_size, void* d_ws, size_t ws_size,
                              hipStream_t stream) {
  (void)in_sizes; (void)n_in; (void)out_size; (void)ws_size;
  const float* state = (const float*)d_in[0];
  const float* mask  = (const float*)d_in[1];
  const float* Wq = (const float*)d_in[2];  const float* bq = (const float*)d_in[3];
  const float* Wk = (const float*)d_in[4];  const float* bk = (const float*)d_in[5];
  const float* Wv = (const float*)d_in[6];  const float* bv = (const float*)d_in[7];
  const float* Whc = (const float*)d_in[8]; const float* bhc = (const float*)d_in[9];
  const float* W1 = (const float*)d_in[10]; const float* b1 = (const float*)d_in[11];
  const float* W2 = (const float*)d_in[12]; const float* b2 = (const float*)d_in[13];
  const float* g1 = (const float*)d_in[14]; const float* be1 = (const float*)d_in[15];
  const float* g2 = (const float*)d_in[16]; const float* be2 = (const float*)d_in[17];
  const float* Wi = (const float*)d_in[18]; const float* bi = (const float*)d_in[19];
  const float* Wf = (const float*)d_in[20]; const float* bf = (const float*)d_in[21];
  const float* Wo = (const float*)d_in[22]; const float* bo = (const float*)d_in[23];
  const float* Wz = (const float*)d_in[24]; const float* bz = (const float*)d_in[25];
  const float* Ri = (const float*)d_in[26]; const float* rbi = (const float*)d_in[27];
  const float* Rf = (const float*)d_in[28]; const float* rbf = (const float*)d_in[29];
  const float* Ro = (const float*)d_in[30]; const float* rbo = (const float*)d_in[31];
  const float* Rz = (const float*)d_in[32]; const float* rbz = (const float*)d_in[33];

  // workspace carve-out (~1.23 GB: persistent state, per-step intermediates,
  // prepped f16 fragment-major weights)
  size_t off = 0;
  char* ws = (char*)d_ws;
  auto carve = [&](size_t bytes) -> void* {
    void* p = ws + off;
    off += (bytes + 255) & ~(size_t)255;
    return p;
  };
  _Float16* hF  = (_Float16*)carve((size_t)ROWS * FF * 2);    // h in f16 (GEMM A)
  float*    Cst = (float*)carve((size_t)ROWS * FF * 4);
  float*    Mst = (float*)carve((size_t)ROWS * FF * 4);
  float*    Nst = (float*)carve((size_t)ROWS * FF * 4);
  _Float16* cat = (_Float16*)carve((size_t)CROWS * DIMP * 2);
  _Float16* qf  = (_Float16*)carve((size_t)CROWS * DIMP * 2);
  _Float16* kf  = (_Float16*)carve((size_t)CROWS * DIMP * 2);
  _Float16* vf  = (_Float16*)carve((size_t)CROWS * DIMP * 2);
  _Float16* z2  = (_Float16*)carve((size_t)ROWS * DIMP * 2);
  _Float16* a1  = (_Float16*)carve((size_t)ROWS * FF * 2);
  _Float16* z5  = (_Float16*)carve((size_t)ROWS * DIMP * 2);
  // prepped weights (f16, fragment-major, padded)
  _Float16* WhcP = (_Float16*)carve((size_t)FF * DIMP * 2);
  _Float16* WqP  = (_Float16*)carve((size_t)DIMP * DIMP * 2);
  _Float16* WkP  = (_Float16*)carve((size_t)DIMP * DIMP * 2);
  _Float16* WvP  = (_Float16*)carve((size_t)DIMP * DIMP * 2);
  _Float16* W1P  = (_Float16*)carve((size_t)DIMP * FF * 2);
  _Float16* W2P  = (_Float16*)carve((size_t)FF * DIMP * 2);
  _Float16* WiP  = (_Float16*)carve((size_t)FF * FF * 2);
  _Float16* WfP  = (_Float16*)carve((size_t)FF * FF * 2);
  _Float16* WoP  = (_Float16*)carve((size_t)FF * FF * 2);
  _Float16* WzP  = (_Float16*)carve((size_t)FF * FF * 2);
  _Float16* RiP  = (_Float16*)carve((size_t)DIMP * FF * 2);
  _Float16* RfP  = (_Float16*)carve((size_t)DIMP * FF * 2);
  _Float16* RoP  = (_Float16*)carve((size_t)DIMP * FF * 2);
  _Float16* RzP  = (_Float16*)carve((size_t)DIMP * FF * 2);
  float*    bsum = (float*)carve((size_t)4 * FF * 4);
  float*    H    = (float*)d_out;  // h lives in d_out across steps

  // --- one-time per-launch weight prep (deterministic) ---
  k_prep_w<<<FF * DIMP / 256, 256, 0, stream>>>(Whc, WhcP, DIMP / 16, FF, DIM);
  k_prep_w<<<DIMP * DIMP / 256, 256, 0, stream>>>(Wq, WqP, DIMP / 16, DIM, DIM);
  k_prep_w<<<DIMP * DIMP / 256, 256, 0, stream>>>(Wk, WkP, DIMP / 16, DIM, DIM);
  k_prep_w<<<DIMP * DIMP / 256, 256, 0, stream>>>(Wv, WvP, DIMP / 16, DIM, DIM);
  k_prep_w<<<DIMP * FF / 256, 256, 0, stream>>>(W1, W1P, FF / 16, DIM, FF);
  k_prep_w<<<FF * DIMP / 256, 256, 0, stream>>>(W2, W2P, DIMP / 16, FF, DIM);
  k_prep_w<<<FF * FF / 256, 256, 0, stream>>>(Wi, WiP, FF / 16, FF, FF);
  k_prep_w<<<FF * FF / 256, 256, 0, stream>>>(Wf, WfP, FF / 16, FF, FF);
  k_prep_w<<<FF * FF / 256, 256, 0, stream>>>(Wo, WoP, FF / 16, FF, FF);
  k_prep_w<<<FF * FF / 256, 256, 0, stream>>>(Wz, WzP, FF / 16, FF, FF);
  k_prep_w<<<DIMP * FF / 256, 256, 0, stream>>>(Ri, RiP, FF / 16, DIM, FF);
  k_prep_w<<<DIMP * FF / 256, 256, 0, stream>>>(Rf, RfP, FF / 16, DIM, FF);
  k_prep_w<<<DIMP * FF / 256, 256, 0, stream>>>(Ro, RoP, FF / 16, DIM, FF);
  k_prep_w<<<DIMP * FF / 256, 256, 0, stream>>>(Rz, RzP, FF / 16, DIM, FF);
  k_prep_bias<<<FF / 256, 256, 0, stream>>>(bi, rbi, bsum + 0 * FF);
  k_prep_bias<<<FF / 256, 256, 0, stream>>>(bf, rbf, bsum + 1 * FF);
  k_prep_bias<<<FF / 256, 256, 0, stream>>>(bo, rbo, bsum + 2 * FF);
  k_prep_bias<<<FF / 256, 256, 0, stream>>>(bz, rbz, bsum + 3 * FF);

  hipMemsetAsync(hF,  0, (size_t)ROWS * FF * 2, stream);
  hipMemsetAsync(Cst, 0, (size_t)ROWS * FF * 4, stream);
  hipMemsetAsync(Mst, 0, (size_t)ROWS * FF * 4, stream);
  hipMemsetAsync(Nst, 0, (size_t)ROWS * FF * 4, stream);
  hipMemsetAsync(H,   0, (size_t)ROWS * FF * 4, stream);

  for (int t = 0; t < SEQL; ++t) {
    k_hc_cat<<<ROWS / 64, 256, 0, stream>>>(hF, WhcP, bhc, state, t, cat);
    k_qkv<<<dim3(CROWS / 64, 3), 256, 0, stream>>>(cat, WqP, bq, WkP, bk,
                                                   WvP, bv, qf, kf, vf);
    k_attn_ln<<<BATCH / 8, 256, 0, stream>>>(qf, kf, vf, state, t, g1, be1, z2);
    k_ffn1<<<dim3(ROWS / 64, FF / 64), 256, 0, stream>>>(z2, W1P, b1, a1);
    k_ffn2_ln<<<ROWS / 64, 256, 0, stream>>>(a1, W2P, b2, z2, g2, be2, z5);
    k_gates<<<dim3(ROWS / 64, FF / 64), 256, 0, stream>>>(
        hF, z5, WiP, WfP, WoP, WzP, RiP, RfP, RoP, RzP,
        bsum, mask, t, Cst, Mst, Nst, H, hF);
  }
}
